// GRU_7619271983664
// MI455X (gfx1250) — matrix operand order repaired and checked by reading, not compiled, not run on hardware
//
#include <hip/hip_runtime.h>
#include <hip/hip_bf16.h>

// ---------------------------------------------------------------------------
// GRU on MI455X (gfx1250): bf16 WMMA 16x16x32, f32 accumulate.
//   Phase 1 (parallel): Gx[m,3072] = X[m,1024] @ [Wxr|Wxz|Wxh] + [br|bz|bh]
//   Phase 2 (serial, 512 steps): fused rz-GEMM then h~-GEMM per step
//   Phase 3: logits = h @ Why + by
// B tiles are staged global->LDS with CDNA5 async copies
// (GLOBAL_LOAD_ASYNC_TO_LDS_B128, ASYNCcnt) and double-buffered so the copy
// of K-tile k+1 overlaps the WMMAs of K-tile k.  A sched_barrier pins all
// four B-fragment DS loads before the WMMA group so the four WMMAs issue
// back-to-back behind a single dscnt wait (critical at 1 wave/SIMD occupancy).
// ---------------------------------------------------------------------------

typedef __bf16 bf16;
typedef __attribute__((ext_vector_type(16))) __bf16 v16bf;
typedef __attribute__((ext_vector_type(8)))  float  v8f;

#define EMB  1024
#define HID  1024
#define OUTV 32000
#define BB   64
#define SS   512

__device__ __forceinline__ void wait_asynccnt0() {
#if __has_builtin(__builtin_amdgcn_s_wait_asynccnt)
  __builtin_amdgcn_s_wait_asynccnt(0);
#else
  asm volatile("s_wait_asynccnt 0x0" ::: "memory");
#endif
}

// Scheduling fence: ALU/VALU/SALU/VMEM may cross (0x77); DS and WMMA may not.
__device__ __forceinline__ void sched_fence_ds_wmma() {
#if __has_builtin(__builtin_amdgcn_sched_barrier)
  __builtin_amdgcn_sched_barrier(0x77);
#endif
}

// Issue two 16B async global->LDS copies (one B-tile chunk pair per thread).
__device__ __forceinline__ void async_copy_pair(unsigned lds0, const bf16* g0) {
  asm volatile(
      "global_load_async_to_lds_b128 %0, %2, off\n\t"
      "global_load_async_to_lds_b128 %1, %3, off"
      :
      : "v"(lds0), "v"(lds0 + 32u), "v"(g0), "v"(g0 + 16)
      : "memory");
}

// ---------------------------------------------------------------------------
// Core: one 128-thread block (4 wave32) computes a 64x64 f32 tile of
// C = A[ M,K ] * Bt[ N,K ]^T.  Wave w owns rows [w*16, w*16+16) x 64 cols.
// A fragments: direct global loads (2 x b128 per lane per K-step).
// B tile (64 rows of Bt x 32 K, 4KB) double-buffered in LDS via async copies.
// ---------------------------------------------------------------------------
__device__ __forceinline__ void wmma_core_16x64(
    const bf16* __restrict__ Aw,   // A + (row base of this wave) * lda
    int lda,
    const bf16* __restrict__ Btb,  // Bt + bN * ldb
    int ldb,
    int K,
    v8f acc[4],
    bf16* sB)                      // shared, 2 buffers x 64*32 bf16 (8KB)
{
  const int lane  = threadIdx.x & 31;
  const int khalf = lane >> 4;     // 0: K 0-7/16-23 ; 1: K 8-15/24-31
  const int rl    = lane & 15;
  const bf16* arow = Aw + (size_t)rl * lda;

  // per-thread slice of the cooperative B-tile copy (128 thr x 32B = 4KB)
  const int row = threadIdx.x >> 1;        // 0..63
  const int q   = threadIdx.x & 1;         // chunk pair selector
  const unsigned ldsbase =
      (unsigned)(uintptr_t)sB + (unsigned)(row * 64 + q * 16);
  const bf16* gbase = Btb + (size_t)row * ldb + q * 8;

  // prologue: stage K-tile 0 into buffer 0
  async_copy_pair(ldsbase, gbase);

  for (int kb = 0; kb < K; kb += 32) {
    const int cur = (kb >> 5) & 1;

    wait_asynccnt0();   // this wave's copies for tile `cur` have landed
    __syncthreads();    // ... and everyone else's too; prev buffer fully read

    if (kb + 32 < K)    // overlap: stage next tile into the other buffer
      async_copy_pair(ldsbase + (unsigned)((1 - cur) * 4096), gbase + kb + 32);

    // A fragment: lane<16 -> row rl, K {kb+0..7, kb+16..23};
    //             lane>=16 -> row rl, K {kb+8..15, kb+24..31}
    v16bf af;
    {
      const bf16* p = arow + kb + khalf * 8;
      ((uint4*)&af)[0] = *(const uint4*)(p);
      ((uint4*)&af)[1] = *(const uint4*)(p + 16);
    }

    // All four B fragments loaded first (one clause of 8 ds_load_b128)...
    const bf16* sT = sB + cur * 2048;
    v16bf bfr[4];
#pragma unroll
    for (int c = 0; c < 4; ++c) {
      // B fragment: lane<16 -> col, K {kb+0..7, kb+16..23} ; lane>=16 -> col, K {kb+8..15, kb+24..31}
      const bf16* qp = sT + (c * 16 + rl) * 32 + khalf * 8;
      ((uint4*)&bfr[c])[0] = ((const uint4*)qp)[0];
      ((uint4*)&bfr[c])[1] = ((const uint4*)qp)[2];
    }

    // ...then a scheduling fence (DS may not sink, WMMA may not hoist), so
    // the four fragments stay simultaneously live in distinct registers and
    // the four WMMAs issue back-to-back behind one dscnt wait.
    sched_fence_ds_wmma();

#pragma unroll
    for (int c = 0; c < 4; ++c) {
      acc[c] = __builtin_amdgcn_wmma_f32_16x16x32_bf16(
          false, af, false, bfr[c], (short)0, acc[c], false, false);
    }
  }
}

// C/D layout: lane<16 -> acc[c][v] = C[v][lane]; lane>=16 -> C[v+8][lane-16]
#define EPILOGUE_FOREACH(BODY)                                   \
  {                                                              \
    const int lane  = threadIdx.x & 31;                          \
    const int khalf = lane >> 4;                                 \
    const int rl    = lane & 15;                                 \
    _Pragma("unroll")                                            \
    for (int c = 0; c < 4; ++c) {                                \
      _Pragma("unroll")                                          \
      for (int v = 0; v < 8; ++v) {                              \
        const int mrow = khalf * 8 + v; /* within wave tile */   \
        const int ncol = c * 16 + rl;   /* within block tile */  \
        BODY                                                     \
      }                                                          \
    }                                                            \
  }

// ---------------------------------------------------------------------------
// Phase 1: Gx = X @ Wxt_cat^T + biascat   (M=32768, N=3072, K=1024) -> bf16
// ---------------------------------------------------------------------------
__global__ __launch_bounds__(128) void k_gemm_gx(
    const bf16* __restrict__ Xbf, const bf16* __restrict__ Wxt,
    const float* __restrict__ biascat, bf16* __restrict__ Gx) {
  __shared__ __align__(128) bf16 sB[2 * 64 * 32];
  const int w  = threadIdx.x >> 5;
  const int bM = blockIdx.x * 64;
  const int bN = blockIdx.y * 64;
  v8f acc[4] = {};
  wmma_core_16x64(Xbf + (size_t)(bM + w * 16) * EMB, EMB,
                  Wxt + (size_t)bN * EMB, EMB, EMB, acc, sB);
  EPILOGUE_FOREACH({
    const int m = bM + w * 16 + mrow;
    const int n = bN + ncol;
    Gx[(size_t)m * 3072 + n] = (bf16)(acc[c][v] + biascat[n]);
  })
}

// ---------------------------------------------------------------------------
// Phase 2a: pre_rz = h @ [Whr|Whz]^T ; r = sigmoid(pre_r + gx_r);
//           z = sigmoid(pre_z + gx_z); store rh = r*h (bf16) and z (f32)
// M=64, N=2048, K=1024 ; grid 32 blocks.
// ---------------------------------------------------------------------------
__global__ __launch_bounds__(128) void k_gemm_rz(
    const bf16* __restrict__ h_bf, const bf16* __restrict__ Whrzt,
    const bf16* __restrict__ gx_t, bf16* __restrict__ rh_bf,
    float* __restrict__ z_f) {
  __shared__ __align__(128) bf16 sB[2 * 64 * 32];
  const int w  = threadIdx.x >> 5;
  const int bN = blockIdx.x * 64;
  v8f acc[4] = {};
  wmma_core_16x64(h_bf + (size_t)(w * 16) * HID, HID,
                  Whrzt + (size_t)bN * HID, HID, HID, acc, sB);
  EPILOGUE_FOREACH({
    const int m = w * 16 + mrow;
    const int n = bN + ncol;                       // 0..2047 (r | z)
    const float pre = acc[c][v] + (float)gx_t[(size_t)m * 3072 + n];
    const float s = 1.0f / (1.0f + __expf(-pre));
    if (n < HID) {
      rh_bf[m * HID + n] = (bf16)(s * (float)h_bf[m * HID + n]);
    } else {
      z_f[m * HID + (n - HID)] = s;
    }
  })
}

// ---------------------------------------------------------------------------
// Phase 2b: h~ = tanh((r*h) @ Whh^T + gx_h) ; h = h + z*(h~ - h)  (in place)
// M=64, N=1024, K=1024 ; grid 16 blocks.
// ---------------------------------------------------------------------------
__global__ __launch_bounds__(128) void k_gemm_h(
    const bf16* __restrict__ rh_bf, const bf16* __restrict__ Whht,
    const bf16* __restrict__ gx_t, const float* __restrict__ z_f,
    bf16* __restrict__ h_bf) {
  __shared__ __align__(128) bf16 sB[2 * 64 * 32];
  const int w  = threadIdx.x >> 5;
  const int bN = blockIdx.x * 64;
  v8f acc[4] = {};
  wmma_core_16x64(rh_bf + (size_t)(w * 16) * HID, HID,
                  Whht + (size_t)bN * HID, HID, HID, acc, sB);
  EPILOGUE_FOREACH({
    const int m = w * 16 + mrow;
    const int n = bN + ncol;
    const float pre = acc[c][v] + (float)gx_t[(size_t)m * 3072 + 2048 + n];
    const float ht  = tanhf(pre);
    const float h   = (float)h_bf[m * HID + n];
    const float z   = z_f[m * HID + n];
    h_bf[m * HID + n] = (bf16)(h + z * (ht - h));  // (1-z)h + z*h~
  })
}

// ---------------------------------------------------------------------------
// Phase 3: out = h @ Why^T + by   (M=64, N=32000, K=1024) -> f32
// ---------------------------------------------------------------------------
__global__ __launch_bounds__(128) void k_gemm_out(
    const bf16* __restrict__ h_bf, const bf16* __restrict__ Whyt,
    const float* __restrict__ by, float* __restrict__ out) {
  __shared__ __align__(128) bf16 sB[2 * 64 * 32];
  const int w  = threadIdx.x >> 5;
  const int bN = blockIdx.x * 64;
  v8f acc[4] = {};
  wmma_core_16x64(h_bf + (size_t)(w * 16) * HID, HID,
                  Whyt + (size_t)bN * HID, HID, HID, acc, sB);
  EPILOGUE_FOREACH({
    const int m = w * 16 + mrow;
    const int n = bN + ncol;
    out[(size_t)m * OUTV + n] = acc[c][v] + by[n];
  })
}

// ---------------------------------------------------------------------------
// Prep kernels
// ---------------------------------------------------------------------------
__global__ void k_tcast(const float* __restrict__ W, bf16* __restrict__ Wt,
                        int K, int N) {
  const size_t idx = (size_t)blockIdx.x * blockDim.x + threadIdx.x;
  if (idx >= (size_t)K * N) return;
  const int n = (int)(idx % N);
  const int k = (int)(idx / N);
  Wt[(size_t)n * K + k] = (bf16)W[idx];   // coalesced read, transposed write
}

__global__ void k_embed(const int* __restrict__ x, const float* __restrict__ emb,
                        bf16* __restrict__ Xbf) {
  const size_t idx = (size_t)blockIdx.x * blockDim.x + threadIdx.x;
  if (idx >= (size_t)SS * BB * EMB) return;
  const int e = (int)(idx & (EMB - 1));
  const size_t m = idx >> 10;              // m = s*B + b
  const int s = (int)(m / BB);
  const int b = (int)(m % BB);
  const int tok = x[(size_t)b * SS + s];
  Xbf[idx] = (bf16)emb[(size_t)tok * EMB + e];
}

__global__ void k_biascat(const float* __restrict__ br, const float* __restrict__ bz,
                          const float* __restrict__ bh, float* __restrict__ bc) {
  const int i = blockIdx.x * blockDim.x + threadIdx.x;
  if (i < HID)            bc[i] = br[i];
  else if (i < 2 * HID)   bc[i] = bz[i - HID];
  else if (i < 3 * HID)   bc[i] = bh[i - 2 * HID];
}

__global__ void k_zero_bf(bf16* __restrict__ p, int n) {
  const int i = blockIdx.x * blockDim.x + threadIdx.x;
  if (i < n) p[i] = (bf16)0.0f;
}

// ---------------------------------------------------------------------------
extern "C" void kernel_launch(void* const* d_in, const int* in_sizes, int n_in,
                              void* d_out, int out_size, void* d_ws, size_t ws_size,
                              hipStream_t stream) {
  const int*   x   = (const int*)  d_in[0];
  const float* emb = (const float*)d_in[1];
  const float* Wxr = (const float*)d_in[2];
  const float* Whr = (const float*)d_in[3];
  const float* br  = (const float*)d_in[4];
  const float* Wxz = (const float*)d_in[5];
  const float* Whz = (const float*)d_in[6];
  const float* bz  = (const float*)d_in[7];
  const float* Wxh = (const float*)d_in[8];
  const float* Whh = (const float*)d_in[9];
  const float* bh  = (const float*)d_in[10];
  const float* Why = (const float*)d_in[11];
  const float* by  = (const float*)d_in[12];

  char* p = (char*)d_ws;
  bf16*  Wxt     = (bf16*)p;  p += (size_t)3 * HID * EMB * 2;      // [3072][1024]
  bf16*  Whrzt   = (bf16*)p;  p += (size_t)2 * HID * HID * 2;      // [2048][1024]
  bf16*  Whht    = (bf16*)p;  p += (size_t)HID * HID * 2;          // [1024][1024]
  bf16*  Whyt    = (bf16*)p;  p += (size_t)OUTV * HID * 2;         // [32000][1024]
  float* biascat = (float*)p; p += (size_t)3 * HID * 4;
  bf16*  h_bf    = (bf16*)p;  p += (size_t)BB * HID * 2;
  bf16*  rh_bf   = (bf16*)p;  p += (size_t)BB * HID * 2;
  float* z_f     = (float*)p; p += (size_t)BB * HID * 4;
  bf16*  Xbf     = (bf16*)p;  p += (size_t)SS * BB * EMB * 2;
  bf16*  Gx      = (bf16*)p;  p += (size_t)SS * BB * 3 * HID * 2;

  // One-time transpose-casts f32 -> bf16 [N][K]
  const size_t sq = (size_t)HID * HID;
  k_tcast<<<(unsigned)((sq + 255) / 256), 256, 0, stream>>>(Wxr, Wxt,                 HID, HID);
  k_tcast<<<(unsigned)((sq + 255) / 256), 256, 0, stream>>>(Wxz, Wxt + sq,            HID, HID);
  k_tcast<<<(unsigned)((sq + 255) / 256), 256, 0, stream>>>(Wxh, Wxt + 2 * sq,        HID, HID);
  k_tcast<<<(unsigned)((sq + 255) / 256), 256, 0, stream>>>(Whr, Whrzt,               HID, HID);
  k_tcast<<<(unsigned)((sq + 255) / 256), 256, 0, stream>>>(Whz, Whrzt + sq,          HID, HID);
  k_tcast<<<(unsigned)((sq + 255) / 256), 256, 0, stream>>>(Whh, Whht,                HID, HID);
  const size_t ysz = (size_t)HID * OUTV;
  k_tcast<<<(unsigned)((ysz + 255) / 256), 256, 0, stream>>>(Why, Whyt, HID, OUTV);

  k_biascat<<<(3 * HID + 255) / 256, 256, 0, stream>>>(br, bz, bh, biascat);

  const size_t xn = (size_t)SS * BB * EMB;
  k_embed<<<(unsigned)((xn + 255) / 256), 256, 0, stream>>>(x, emb, Xbf);

  k_zero_bf<<<(BB * HID + 255) / 256, 256, 0, stream>>>(h_bf, BB * HID);

  // Phase 1: parallel input-projection GEMM (512 x 48 blocks)
  dim3 g1(SS * BB / 64, 3 * HID / 64);
  k_gemm_gx<<<g1, 128, 0, stream>>>(Xbf, Wxt, biascat, Gx);

  // Phase 2: 512 sequential steps, 2 fused GEMMs each (graph-captured)
  for (int t = 0; t < SS; ++t) {
    const bf16* gx_t = Gx + (size_t)t * BB * 3 * HID;
    k_gemm_rz<<<2 * HID / 64, 128, 0, stream>>>(h_bf, Whrzt, gx_t, rh_bf, z_f);
    k_gemm_h <<<    HID / 64, 128, 0, stream>>>(rh_bf, Whht, gx_t, z_f, h_bf);
  }

  // Phase 3: logits
  k_gemm_out<<<OUTV / 64, 128, 0, stream>>>(h_bf, Whyt, by, (float*)d_out);
}
